// TimeSeriesLSTMRegressor_74062416052836
// MI455X (gfx1250) — matrix operand order repaired and checked by reading, not compiled, not run on hardware
//
#include <hip/hip_runtime.h>

// Fused 2-layer LSTM (B=4096, T=512, IN=16, H=32) + MLP head for gfx1250.
// Per block: 16 batch rows, 8 waves; each wave owns a 16-wide gate column
// slice and issues v_wmma_f32_16x16x32_f16 ops with weight B-fragments
// resident in VGPRs for all 512 timesteps. h/c recurrent state never
// touches HBM: only x (128MB, compulsory) is streamed in.
//
// Round-2: activations via hardware TRANS ops (v_rcp_f32 / v_tanh_f32).
// Round-3/4: software-pipeline the x global load by one timestep so the HBM
// latency wait sits at the END of the loop body (hidden behind both WMMA
// phases, elementwise, and barriers); x loads are non-temporal via clang
// ext_vector float4 (HIP's float4 struct is rejected by the builtin).

typedef __attribute__((ext_vector_type(16))) _Float16 v16h;
typedef __attribute__((ext_vector_type(8)))  _Float16 v8h;
typedef __attribute__((ext_vector_type(8)))  float    v8f;
typedef __attribute__((ext_vector_type(4)))  float    v4f;

#define TSTEPS  512
#define INDIM   16
#define HDIM    32
#define BTILE   16
#define GSTRIDE 132   // 128 gate cols + 4 pad floats (LDS bank spread)

__device__ __forceinline__ float rcp_(float x) { return __builtin_amdgcn_rcpf(x); }
__device__ __forceinline__ float sigm_(float x) { return rcp_(1.f + __expf(-x)); }
#if __has_builtin(__builtin_amdgcn_tanhf)
__device__ __forceinline__ float tanh_(float x) { return __builtin_amdgcn_tanhf(x); }
#else
__device__ __forceinline__ float tanh_(float x) { return 1.f - 2.f * rcp_(__expf(2.f * x) + 1.f); }
#endif

// Build 16-bit A-fragment for one 16x32 tile from row-major f16 LDS.
// ISA A layout (16-bit, 16x32): lane<16 -> M=lane, halves 0..7 = K 0..7,
// halves 8..15 = K 16..23; lane>=16 -> M=lane-16, K 8..15 / 24..31.
__device__ __forceinline__ v16h load_ah(const _Float16* hs, int am, int koff) {
  const v8h* p0 = reinterpret_cast<const v8h*>(hs + am * HDIM + koff);
  const v8h* p1 = reinterpret_cast<const v8h*>(hs + am * HDIM + koff + 16);
  v8h lo = *p0, hi = *p1;
  v16h a;
#pragma unroll
  for (int i = 0; i < 8; ++i) { a[i] = lo[i]; a[i + 8] = hi[i]; }
  return a;
}

extern "C" __global__ __launch_bounds__(256)
void lstm2_fused(const float* __restrict__ x,
                 const float* __restrict__ Wih0, const float* __restrict__ Whh0,
                 const float* __restrict__ bih0, const float* __restrict__ bhh0,
                 const float* __restrict__ Wih1, const float* __restrict__ Whh1,
                 const float* __restrict__ bih1, const float* __restrict__ bhh1,
                 const float* __restrict__ W1,   const float* __restrict__ b1,
                 const float* __restrict__ W2,   const float* __restrict__ b2,
                 float* __restrict__ out)
{
  __shared__ float    gates[BTILE * GSTRIDE];  // 8448 B, reused by both layers
  __shared__ _Float16 h0s[BTILE * HDIM];       // layer-0 hidden (f16, row major)
  __shared__ _Float16 h2s[BTILE * HDIM];       // layer-1 hidden
  __shared__ float    zbuf[BTILE * 16];        // head intermediate

  const int tid  = threadIdx.x;
  const int lane = tid & 31;
  const int w    = tid >> 5;            // wave id 0..7 -> gate n-tile
  const int b0   = blockIdx.x * BTILE;  // batch tile origin
  const int ncol = w * 16;
  const int nloc = lane & 15;           // C/D layout: N = lane & 15

  // ---- persistent f16 B fragments (B[k][n] = W[n][k]; N = lane & 15, half p = K in table order) ----
  v16h B0x, B0h, B1x, B1h;
#pragma unroll
  for (int p = 0; p < 16; ++p) {
    const int n = ncol + (lane & 15);
    const int kk = (p < 8) ? ((lane >> 4) * 8 + p) : (16 + (lane >> 4) * 8 + (p - 8));
    B0x[p] = (kk < INDIM) ? (_Float16)Wih0[n * INDIM + kk] : (_Float16)0.f; // K 16..31 zero-pad
    B0h[p] = (_Float16)Whh0[n * HDIM + kk];
    B1x[p] = (_Float16)Wih1[n * HDIM + kk];
    B1h[p] = (_Float16)Whh1[n * HDIM + kk];
  }
  const float bias0 = bih0[ncol + nloc] + bhh0[ncol + nloc];
  const float bias1 = bih1[ncol + nloc] + bhh1[ncol + nloc];

  // ---- zero recurrent state ----
  h0s[tid] = (_Float16)0.f; h0s[tid + 256] = (_Float16)0.f;
  h2s[tid] = (_Float16)0.f; h2s[tid + 256] = (_Float16)0.f;
  __syncthreads();

  // elementwise cell ownership: 2 (m,j) cells per thread, c in registers
  const int mA = tid >> 5,          jA = tid & 31;
  const int mB = (tid + 256) >> 5,  jB = tid & 31;
  float c0a = 0.f, c0b = 0.f, c1a = 0.f, c1b = 0.f;

  // A-fragment addressing for this lane
  const int am   = lane & 15;
  const int koff = (lane < 16) ? 0 : 8;
  const float* xbase = x + ((size_t)(b0 + am) * TSTEPS) * INDIM + koff;
  const int gm = (lane < 16) ? 0 : 8;
  const int gn = ncol + nloc;

  // ---- prime the x pipeline (t = 0) ----
  v4f xv0 = __builtin_nontemporal_load(reinterpret_cast<const v4f*>(xbase));
  v4f xv1 = __builtin_nontemporal_load(reinterpret_cast<const v4f*>(xbase + 4));

#pragma unroll 1
  for (int t = 0; t < TSTEPS; ++t) {
    // -- issue next step's x load now; its wait lands at the loop tail --
    const int tn = (t + 1 < TSTEPS) ? t + 1 : t;   // scalar clamp (redundant last load)
    const v4f* xpn = reinterpret_cast<const v4f*>(xbase + (size_t)tn * INDIM);
    v4f nx0 = __builtin_nontemporal_load(xpn);
    v4f nx1 = __builtin_nontemporal_load(xpn + 1);

    // -- A fragments: x from pipelined registers, h from LDS (prev step) --
    v16h Ax;
#pragma unroll
    for (int i = 0; i < 4; ++i) { Ax[i] = (_Float16)xv0[i]; Ax[i + 4] = (_Float16)xv1[i]; }
#pragma unroll
    for (int i = 8; i < 16; ++i) Ax[i] = (_Float16)0.f;   // K>=16 pad (IN=16)
    v16h Ah0 = load_ah(h0s, am, koff);
    v16h Ah2 = load_ah(h2s, am, koff);

    // -- layer 0 gates: [x|pad]@Wih0^T + h0@Whh0^T + b --
    v8f acc;
#pragma unroll
    for (int r = 0; r < 8; ++r) acc[r] = bias0;
    acc = __builtin_amdgcn_wmma_f32_16x16x32_f16(false, Ax,  false, B0x, (short)0, acc, false, false);
    acc = __builtin_amdgcn_wmma_f32_16x16x32_f16(false, Ah0, false, B0h, (short)0, acc, false, false);
#pragma unroll
    for (int r = 0; r < 8; ++r) gates[(gm + r) * GSTRIDE + gn] = acc[r];
    __syncthreads();  // B1: gates ready; prior h0s/h2s reads complete

    // -- layer 0 elementwise (i,f,g,o at col j, j+32, j+64, j+96) --
    {
      float i0 = gates[mA * GSTRIDE + jA],      f0 = gates[mA * GSTRIDE + 32 + jA];
      float g0 = gates[mA * GSTRIDE + 64 + jA], o0 = gates[mA * GSTRIDE + 96 + jA];
      c0a = sigm_(f0) * c0a + sigm_(i0) * tanh_(g0);
      h0s[mA * HDIM + jA] = (_Float16)(sigm_(o0) * tanh_(c0a));
      float i1 = gates[mB * GSTRIDE + jB],      f1 = gates[mB * GSTRIDE + 32 + jB];
      float g1 = gates[mB * GSTRIDE + 64 + jB], o1 = gates[mB * GSTRIDE + 96 + jB];
      c0b = sigm_(f1) * c0b + sigm_(i1) * tanh_(g1);
      h0s[mB * HDIM + jB] = (_Float16)(sigm_(o1) * tanh_(c0b));
    }
    __syncthreads();  // B2: h0s ready; gate reads done

    // -- layer 1 gates: h1@Wih1^T + h2@Whh1^T + b --
    v16h Ah1 = load_ah(h0s, am, koff);
#pragma unroll
    for (int r = 0; r < 8; ++r) acc[r] = bias1;
    acc = __builtin_amdgcn_wmma_f32_16x16x32_f16(false, Ah1, false, B1x, (short)0, acc, false, false);
    acc = __builtin_amdgcn_wmma_f32_16x16x32_f16(false, Ah2, false, B1h, (short)0, acc, false, false);
#pragma unroll
    for (int r = 0; r < 8; ++r) gates[(gm + r) * GSTRIDE + gn] = acc[r];
    __syncthreads();  // B3

    // -- layer 1 elementwise --
    {
      float i0 = gates[mA * GSTRIDE + jA],      f0 = gates[mA * GSTRIDE + 32 + jA];
      float g0 = gates[mA * GSTRIDE + 64 + jA], o0 = gates[mA * GSTRIDE + 96 + jA];
      c1a = sigm_(f0) * c1a + sigm_(i0) * tanh_(g0);
      h2s[mA * HDIM + jA] = (_Float16)(sigm_(o0) * tanh_(c1a));
      float i1 = gates[mB * GSTRIDE + jB],      f1 = gates[mB * GSTRIDE + 32 + jB];
      float g1 = gates[mB * GSTRIDE + 64 + jB], o1 = gates[mB * GSTRIDE + 96 + jB];
      c1b = sigm_(f1) * c1b + sigm_(i1) * tanh_(g1);
      h2s[mB * HDIM + jB] = (_Float16)(sigm_(o1) * tanh_(c1b));
    }
    __syncthreads();  // B4: h2s ready for next step

    // -- rotate pipelined x registers (s_wait_loadcnt materializes here) --
    xv0 = nx0; xv1 = nx1;
  }

  // ---- head: relu(last @ W1^T + b1) @ W2^T + b2 ----
  {
    const int m = tid >> 4, k = tid & 15;   // 16 rows x 16 hidden = 256 threads
    float z = b1[k];
#pragma unroll
    for (int j = 0; j < HDIM; ++j) z += (float)h2s[m * HDIM + j] * W1[k * HDIM + j];
    zbuf[m * 16 + k] = fmaxf(z, 0.f);
  }
  __syncthreads();
  if (tid < BTILE) {
    float s = b2[0];
#pragma unroll
    for (int k = 0; k < 16; ++k) s += zbuf[tid * 16 + k] * W2[k];
    out[b0 + tid] = s;
  }
}

extern "C" void kernel_launch(void* const* d_in, const int* in_sizes, int n_in,
                              void* d_out, int out_size, void* d_ws, size_t ws_size,
                              hipStream_t stream) {
  (void)in_sizes; (void)n_in; (void)d_ws; (void)ws_size; (void)out_size;
  const float* x    = (const float*)d_in[0];
  const float* Wih0 = (const float*)d_in[1];
  const float* Whh0 = (const float*)d_in[2];
  const float* bih0 = (const float*)d_in[3];
  const float* bhh0 = (const float*)d_in[4];
  const float* Wih1 = (const float*)d_in[5];
  const float* Whh1 = (const float*)d_in[6];
  const float* bih1 = (const float*)d_in[7];
  const float* bhh1 = (const float*)d_in[8];
  const float* W1   = (const float*)d_in[9];
  const float* b1   = (const float*)d_in[10];
  const float* W2   = (const float*)d_in[11];
  const float* b2   = (const float*)d_in[12];
  float* out = (float*)d_out;

  // 4096 batch rows / 16 rows per block = 256 blocks, 256 threads (8 waves)
  lstm2_fused<<<256, 256, 0, stream>>>(x, Wih0, Whh0, bih0, bhh0,
                                       Wih1, Whh1, bih1, bhh1,
                                       W1, b1, W2, b2, out);
}
